// RSNN_15668040696086
// MI455X (gfx1250) — matrix operand order repaired and checked by reading, not compiled, not run on hardware
//
#include <hip/hip_runtime.h>
#include <hip/hip_bf16.h>
#include <stdint.h>

typedef __attribute__((ext_vector_type(16))) _Float16 v16h;
typedef __attribute__((ext_vector_type(8)))  _Float16 v8h;
typedef __attribute__((ext_vector_type(8)))  float    v8f;

#define B_   256
#define T_   50
#define I_   1024
#define H_   2048
#define O_   20
#define THRESH 0.3f

// ---------------------------------------------------------------------------
// Prep kernels
// ---------------------------------------------------------------------------
__global__ void k_zero_u32(uint32_t* __restrict__ p, int n) {
    int i = blockIdx.x * blockDim.x + threadIdx.x;
    int stride = gridDim.x * blockDim.x;
    for (; i < n; i += stride) p[i] = 0u;
}

__global__ void k_cvt_f16(_Float16* __restrict__ dst, const float* __restrict__ src, int n) {
    int i = blockIdx.x * blockDim.x + threadIdx.x;
    int stride = gridDim.x * blockDim.x;
    for (; i < n; i += stride) dst[i] = (_Float16)src[i];
}

__global__ void k_alpha(float* __restrict__ dst, const float* __restrict__ tau, int n) {
    int i = blockIdx.x * blockDim.x + threadIdx.x;
    if (i < n) dst[i] = __expf(-1.0f / tau[i]);
}

// ---------------------------------------------------------------------------
// Fragment loaders (contiguous-k VGPR layouts used here, wave32)
// A 16x32 f16: lane L<16 -> M=L, K = 0..15 ; lane L>=16 -> K = 16..31
// B 32x16 f16 (B = W^T): lane L<16 -> N=L, K=0..15 ; L>=16 -> K=16..31  (contig in W row)
// ---------------------------------------------------------------------------
__device__ __forceinline__ v16h load_a_f16(const _Float16* __restrict__ rowbase,
                                           int k0, int lhi) {
    const _Float16* r = rowbase + k0 + lhi * 16;
    v8h lo8 = *(const v8h*)(r);
    v8h hi8 = *(const v8h*)(r + 8);
    return __builtin_shufflevector(lo8, hi8,
        0, 1, 2, 3, 4, 5, 6, 7, 8, 9, 10, 11, 12, 13, 14, 15);
}

// ---------------------------------------------------------------------------
// Hidden-layer step: h_mem = x_t@Wih^T + spk_prev@Whh^T + h_mem*alpha*(1-spk_prev)
// Grid: 32 blocks (each owns 64 cols), 256 threads = 8 waves.
// Wave w owns rows [32w, 32w+32) as two 16-row WMMA tiles x four 16-col tiles.
// ---------------------------------------------------------------------------
__global__ __launch_bounds__(256) void k_hidden(
    const _Float16* __restrict__ inp_h,     // [B][T][I] f16 (precvt)
    int t,
    const _Float16* __restrict__ wih,       // [H][I] f16
    const _Float16* __restrict__ whh,       // [H][H] f16
    const float*    __restrict__ alpha_h,   // [H]
    float*          __restrict__ hmem,      // [B][H]
    const _Float16* __restrict__ spk_prev,  // [B][H] f16 (0/1)
    _Float16*       __restrict__ spk_new)   // [B][H] f16
{
    const int lane  = threadIdx.x & 31;
    const int wave  = threadIdx.x >> 5;
    const int mbase = wave * 32;
    const int wgN   = blockIdx.x * 64;
    const int lhi   = lane >> 4;    // 0 or 1 (K-half select)
    const int llo   = lane & 15;    // row/col within tile

    v8f c[2][4] = {};

    const _Float16* arow0_p1 = inp_h + ((size_t)(mbase + llo) * T_ + t) * I_;
    const _Float16* arow1_p1 = inp_h + ((size_t)(mbase + 16 + llo) * T_ + t) * I_;
    const _Float16* arow0_p2 = spk_prev + (size_t)(mbase + llo) * H_;
    const _Float16* arow1_p2 = spk_prev + (size_t)(mbase + 16 + llo) * H_;
    const _Float16* brow_ih  = wih + (size_t)(wgN + llo) * I_ + lhi * 16;
    const _Float16* brow_hh  = whh + (size_t)(wgN + llo) * H_ + lhi * 16;

    // ---- Part 1: input GEMM, K = 1024 ----
#pragma unroll 2
    for (int k0 = 0; k0 < I_; k0 += 32) {
        v16h a0 = load_a_f16(arow0_p1, k0, lhi);
        v16h a1 = load_a_f16(arow1_p1, k0, lhi);
#pragma unroll
        for (int nj = 0; nj < 4; ++nj) {
            v16h b = *(const v16h*)(brow_ih + (size_t)nj * 16 * I_ + k0);
            c[0][nj] = __builtin_amdgcn_wmma_f32_16x16x32_f16(
                false, a0, false, b, (short)0, c[0][nj], false, false);
            c[1][nj] = __builtin_amdgcn_wmma_f32_16x16x32_f16(
                false, a1, false, b, (short)0, c[1][nj], false, false);
        }
    }

    // ---- Part 2: recurrent GEMM, K = 2048 ----
#pragma unroll 2
    for (int k0 = 0; k0 < H_; k0 += 32) {
        v16h a0 = load_a_f16(arow0_p2, k0, lhi);
        v16h a1 = load_a_f16(arow1_p2, k0, lhi);
#pragma unroll
        for (int nj = 0; nj < 4; ++nj) {
            v16h b = *(const v16h*)(brow_hh + (size_t)nj * 16 * H_ + k0);
            c[0][nj] = __builtin_amdgcn_wmma_f32_16x16x32_f16(
                false, a0, false, b, (short)0, c[0][nj], false, false);
            c[1][nj] = __builtin_amdgcn_wmma_f32_16x16x32_f16(
                false, a1, false, b, (short)0, c[1][nj], false, false);
        }
    }

    // ---- Epilogue: membrane update + spike ----
    // C/D layout: lane L -> N = L&15; VGPR r -> M = r + 8*(L>>4)
#pragma unroll
    for (int mi = 0; mi < 2; ++mi) {
#pragma unroll
        for (int nj = 0; nj < 4; ++nj) {
            const int n = wgN + nj * 16 + llo;
            const float an = alpha_h[n];
#pragma unroll
            for (int r = 0; r < 8; ++r) {
                const int m = mbase + mi * 16 + r + lhi * 8;
                const size_t idx = (size_t)m * H_ + n;
                float prev = (float)spk_prev[idx];
                float hm = c[mi][nj][r] + hmem[idx] * an * (1.0f - prev);
                hmem[idx] = hm;
                spk_new[idx] = (_Float16)(((hm - THRESH) > 0.0f) ? 1.0f : 0.0f);
            }
        }
    }
}

// ---------------------------------------------------------------------------
// Output-layer step: one wave per batch row.
// ---------------------------------------------------------------------------
__global__ __launch_bounds__(256) void k_output(
    const _Float16* __restrict__ spk,      // [B][H] (this step's hidden spikes)
    const float*    __restrict__ who,      // [O][H] f32
    const float*    __restrict__ alpha_o,  // [O]
    float*          __restrict__ omem,     // [B][O]
    float*          __restrict__ ospk,     // [B][O]
    float*          __restrict__ out)      // [2*B*O]: o_sum/T then mot
{
    const int lane = threadIdx.x & 31;
    const int m = (blockIdx.x * blockDim.x + threadIdx.x) >> 5;  // row (0..255)

    float part[O_];
#pragma unroll
    for (int o = 0; o < O_; ++o) part[o] = 0.0f;

    const _Float16* row = spk + (size_t)m * H_;
    for (int k = lane; k < H_; k += 32) {
        float s = (float)row[k];
        if (s != 0.0f) {
#pragma unroll
            for (int o = 0; o < O_; ++o) part[o] += who[o * H_ + k];
        }
    }
    // butterfly reduce: all lanes end with the full sums
#pragma unroll
    for (int off = 16; off >= 1; off >>= 1) {
#pragma unroll
        for (int o = 0; o < O_; ++o)
            part[o] += __shfl_xor(part[o], off, 32);
    }

    float om[O_], e[O_];
    float mx = -1e30f;
#pragma unroll
    for (int o = 0; o < O_; ++o) {
        float pm = omem[m * O_ + o];
        float ps = ospk[m * O_ + o];
        om[o] = pm * alpha_o[o] * (1.0f - ps) + part[o];
        mx = fmaxf(mx, om[o]);
    }
    float se = 0.0f;
#pragma unroll
    for (int o = 0; o < O_; ++o) { e[o] = __expf(om[o] - mx); se += e[o]; }
    const float inv_se = 1.0f / se;

    if (lane == 0) {
#pragma unroll
        for (int o = 0; o < O_; ++o) {
            float sp = ((om[o] - THRESH) > 0.0f) ? 1.0f : 0.0f;
            omem[m * O_ + o] = om[o];
            ospk[m * O_ + o] = sp;
            out[m * O_ + o]           += sp * (1.0f / (float)T_);
            out[B_ * O_ + m * O_ + o] += e[o] * inv_se;
        }
    }
}

// ---------------------------------------------------------------------------
// Host-side launch
// ---------------------------------------------------------------------------
extern "C" void kernel_launch(void* const* d_in, const int* in_sizes, int n_in,
                              void* d_out, int out_size, void* d_ws, size_t ws_size,
                              hipStream_t stream) {
    const float* inp   = (const float*)d_in[0];   // [256][50][1024]
    const float* w_ih  = (const float*)d_in[1];   // [2048][1024]
    const float* w_hh  = (const float*)d_in[2];   // [2048][2048]
    const float* w_ho  = (const float*)d_in[3];   // [20][2048]
    const float* tau_h = (const float*)d_in[4];   // [2048]
    const float* tau_o = (const float*)d_in[5];   // [20]
    float* out = (float*)d_out;

    // workspace carve (256B aligned chunks)
    char* ws = (char*)d_ws;
    size_t off = 0;
    auto carve = [&](size_t bytes) -> char* {
        char* p = ws + off;
        off += (bytes + 255) & ~(size_t)255;
        return p;
    };
    _Float16* wih_h  = (_Float16*)carve((size_t)H_ * I_ * 2);          // 4 MB
    _Float16* whh_h  = (_Float16*)carve((size_t)H_ * H_ * 2);          // 8 MB
    _Float16* inp_h  = (_Float16*)carve((size_t)B_ * T_ * I_ * 2);     // 26 MB
    char* state_base = ws + off;
    float*    hmem   = (float*)   carve((size_t)B_ * H_ * 4);          // 2 MB
    _Float16* hspk0  = (_Float16*)carve((size_t)B_ * H_ * 2);          // 1 MB
    _Float16* hspk1  = (_Float16*)carve((size_t)B_ * H_ * 2);          // 1 MB
    float*    omem   = (float*)   carve((size_t)B_ * O_ * 4);
    float*    ospk   = (float*)   carve((size_t)B_ * O_ * 4);
    size_t state_bytes = (size_t)((ws + off) - state_base);
    float*    alpha_h = (float*)carve((size_t)H_ * 4);
    float*    alpha_o = (float*)carve((size_t)O_ * 4);

    // prep: weight + input conversion, alphas, zero state + output accumulators
    k_cvt_f16<<<1024, 256, 0, stream>>>(wih_h, w_ih, H_ * I_);
    k_cvt_f16<<<1024, 256, 0, stream>>>(whh_h, w_hh, H_ * H_);
    k_cvt_f16<<<2048, 256, 0, stream>>>(inp_h, inp, B_ * T_ * I_);
    k_alpha<<<(H_ + 255) / 256, 256, 0, stream>>>(alpha_h, tau_h, H_);
    k_alpha<<<1, 32, 0, stream>>>(alpha_o, tau_o, O_);
    k_zero_u32<<<1024, 256, 0, stream>>>((uint32_t*)state_base, (int)(state_bytes / 4));
    k_zero_u32<<<(2 * B_ * O_ + 255) / 256, 256, 0, stream>>>((uint32_t*)out, 2 * B_ * O_);

    // time loop: per-step kernels (stream order provides the recurrence sync)
    for (int t = 0; t < T_; ++t) {
        _Float16* spk_prev = (t & 1) ? hspk1 : hspk0;
        _Float16* spk_new  = (t & 1) ? hspk0 : hspk1;
        k_hidden<<<H_ / 64, 256, 0, stream>>>(inp_h, t, wih_h, whh_h, alpha_h,
                                              hmem, spk_prev, spk_new);
        k_output<<<B_ / 8, 256, 0, stream>>>(spk_new, w_ho, alpha_o,
                                             omem, ospk, out);
    }

    (void)in_sizes; (void)n_in; (void)out_size; (void)ws_size;
}
